// EdgeModel_32169305047409
// MI455X (gfx1250) — matrix operand order repaired and checked by reading, not compiled, not run on hardware
//
#include <hip/hip_runtime.h>

typedef __attribute__((ext_vector_type(16))) _Float16 v16h;
typedef __attribute__((ext_vector_type(8)))  _Float16 v8h;
typedef __attribute__((ext_vector_type(8)))  float    v8f;

#define EDGE_IN   16
#define NODE_IN   32
#define COND_IN   16
#define EDGE_OUT  16
#define HIDDEN    64
#define IN_DIM    96

#define WAVES_PER_BLOCK 8
#define BLOCK_THREADS   (WAVES_PER_BLOCK * 32)
#define LDS_PITCH       (HIDDEN + 8)   // 72 halves = 144 B rows -> 16B aligned
#define TILES_PER_WAVE  16

// pack two float4 (8 consecutive K values) into halves [off..off+7] of an A fragment
#define PACK8(Av, off, f0, f1)                                        \
  do {                                                                \
    Av[(off)+0] = (_Float16)(f0).x; Av[(off)+1] = (_Float16)(f0).y;   \
    Av[(off)+2] = (_Float16)(f0).z; Av[(off)+3] = (_Float16)(f0).w;   \
    Av[(off)+4] = (_Float16)(f1).x; Av[(off)+5] = (_Float16)(f1).y;   \
    Av[(off)+6] = (_Float16)(f1).z; Av[(off)+7] = (_Float16)(f1).w;   \
  } while (0)

static __device__ __forceinline__ v8f wmma_f16(v16h a, v16h b, v8f c) {
  // D = A(16x32 f16) * B(32x16 f16) + C(16x16 f32)
  return __builtin_amdgcn_wmma_f32_16x16x32_f16(
      /*neg_a=*/false, a, /*neg_b=*/false, b,
      /*c_mod=*/(short)0, c, /*reuse_a=*/false, /*reuse_b=*/false);
}

__global__ __launch_bounds__(BLOCK_THREADS)
void edge_mlp_wmma(const float* __restrict__ src,
                   const float* __restrict__ dst,
                   const float* __restrict__ ea,
                   const float* __restrict__ u,
                   const int*   __restrict__ batch,
                   const float* __restrict__ W1,
                   const float* __restrict__ b1,
                   const float* __restrict__ W2,
                   const float* __restrict__ b2,
                   float* __restrict__ out,
                   int E, int n_tiles)
{
  // per-wave 16x64 f16 transpose staging for h (no cross-wave sharing)
  __shared__ _Float16 lds_h[WAVES_PER_BLOCK][16][LDS_PITCH];
  // block-shared transposed f16 weights: wt1[n][k] = W1[k][n], wt2[n][k] = W2[k][n]
  __shared__ _Float16 wt1[HIDDEN][IN_DIM];      // 64 x 96, rows 192 B (16B aligned)
  __shared__ _Float16 wt2[EDGE_OUT][HIDDEN];    // 16 x 64, rows 128 B
  __shared__ float    lb1[HIDDEN];
  __shared__ float    lb2[EDGE_OUT];

  // ---- cooperative, coalesced weight staging (once per block) ------------
  for (int i = threadIdx.x; i < IN_DIM * HIDDEN; i += BLOCK_THREADS) {
    const int k = i / HIDDEN, n = i % HIDDEN;
    wt1[n][k] = (_Float16)W1[i];
  }
  for (int i = threadIdx.x; i < HIDDEN * EDGE_OUT; i += BLOCK_THREADS) {
    const int k = i / EDGE_OUT, n = i % EDGE_OUT;
    wt2[n][k] = (_Float16)W2[i];
  }
  if (threadIdx.x < HIDDEN)   lb1[threadIdx.x] = b1[threadIdx.x];
  if (threadIdx.x < EDGE_OUT) lb2[threadIdx.x] = b2[threadIdx.x];
  __syncthreads();

  const int lane = threadIdx.x & 31;
  const int wave = threadIdx.x >> 5;
  const int col  = lane & 15;     // column (N) for B/C/D layouts, row (M) for A reads
  const int kq   = lane >> 4;     // half-wave selector in A/B K-striping

  // ---- constant B fragments, two 8-half LDS reads each ----------------------
  // B 32x16 f16 layout: lane holds column n=col; halves j<8 -> K = kq*8 + j, j>=8 -> K = 16 + kq*8 + (j-8)
  v16h bw1[4][3];                 // [N-tile 0..3][K-tile 0..2] of W1 (96x64)
  #pragma unroll
  for (int nt = 0; nt < 4; ++nt)
    #pragma unroll
    for (int kt = 0; kt < 3; ++kt)
    {
      v8h lo = *(const v8h*)(&wt1[nt * 16 + col][kt * 32 + kq * 8]);
      v8h hi = *(const v8h*)(&wt1[nt * 16 + col][kt * 32 + 16 + kq * 8]);
      bw1[nt][kt] = __builtin_shufflevector(lo, hi, 0,1,2,3,4,5,6,7,8,9,10,11,12,13,14,15);
    }

  v16h bw2[2];                    // [K-tile 0..1] of W2 (64x16)
  #pragma unroll
  for (int kt = 0; kt < 2; ++kt)
  {
    v8h lo = *(const v8h*)(&wt2[col][kt * 32 + kq * 8]);
    v8h hi = *(const v8h*)(&wt2[col][kt * 32 + 16 + kq * 8]);
    bw2[kt] = __builtin_shufflevector(lo, hi, 0,1,2,3,4,5,6,7,8,9,10,11,12,13,14,15);
  }

  float b1v[4];
  #pragma unroll
  for (int nt = 0; nt < 4; ++nt) b1v[nt] = lb1[nt * 16 + col];
  const float b2v = lb2[col];

  const int waves_total = (gridDim.x * BLOCK_THREADS) >> 5;
  int wave_id = (int)((blockIdx.x * BLOCK_THREADS + threadIdx.x) >> 5);

  for (int tile = wave_id; tile < n_tiles; tile += waves_total) {
    const int base = tile * 16;
    int row = base + col;
    if (row >= E) row = E - 1;           // clamp: WMMA needs all lanes active
    const size_t r = (size_t)row;

    // prefetch next tile's node features into the cache hierarchy
    {
      int nt_tile = tile + waves_total;
      if (nt_tile < n_tiles) {
        size_t nrow = (size_t)(nt_tile * 16 + col);
        if (nrow >= (size_t)E) nrow = (size_t)E - 1;
        __builtin_prefetch(src + nrow * NODE_IN, 0, 1);
        __builtin_prefetch(dst + nrow * NODE_IN, 0, 1);
      }
    }

    // ---- build three 16x32 f16 A fragments (K = 0..95 of x) --------------
    // A layout: lanes 0-15 -> K = [0..7]+[16..23]; lanes 16-31 -> +8 in each half
    const float4* s_lo = (const float4*)(src + r * NODE_IN + kq * 8);
    const float4* s_hi = (const float4*)(src + r * NODE_IN + 16 + kq * 8);
    const float4* d_lo = (const float4*)(dst + r * NODE_IN + kq * 8);
    const float4* d_hi = (const float4*)(dst + r * NODE_IN + 16 + kq * 8);
    float4 sl0 = s_lo[0], sl1 = s_lo[1], sh0 = s_hi[0], sh1 = s_hi[1];
    float4 dl0 = d_lo[0], dl1 = d_lo[1], dh0 = d_hi[0], dh1 = d_hi[1];

    const float4* e4 = (const float4*)(ea + r * EDGE_IN + kq * 8);
    float4 e0 = e4[0], e1 = e4[1];

    const int bi = batch[row];
    const float4* u4 = (const float4*)(u + (size_t)bi * COND_IN + kq * 8);
    float4 u0 = u4[0], u1 = u4[1];

    v16h a0, a1, a2;
    PACK8(a0, 0, sl0, sl1); PACK8(a0, 8, sh0, sh1);   // K  0..31 : src
    PACK8(a1, 0, dl0, dl1); PACK8(a1, 8, dh0, dh1);   // K 32..63 : dest
    PACK8(a2, 0, e0,  e1 ); PACK8(a2, 8, u0,  u1 );   // K 64..79 ea, 80..95 u

    // ---- layer 1: h = relu(x @ W1 + b1), 12 WMMAs ------------------------
    v8f acc[4];
    #pragma unroll
    for (int nt = 0; nt < 4; ++nt) {
      v8f c = {};
      c = wmma_f16(a0, bw1[nt][0], c);
      c = wmma_f16(a1, bw1[nt][1], c);
      c = wmma_f16(a2, bw1[nt][2], c);
      acc[nt] = c;
    }

    // bias + relu, stage h (f16) into LDS in row-major [m][n]
    // D layout: VGPR rr -> M = kq*8 + rr, N = col (per N-tile)
    #pragma unroll
    for (int nt = 0; nt < 4; ++nt) {
      #pragma unroll
      for (int rr = 0; rr < 8; ++rr) {
        float v = acc[nt][rr] + b1v[nt];
        v = v > 0.0f ? v : 0.0f;
        lds_h[wave][kq * 8 + rr][nt * 16 + col] = (_Float16)v;
      }
    }

    // ---- re-read h in A layout (in-wave transpose through LDS) -----------
    v16h ha[2];
    #pragma unroll
    for (int t = 0; t < 2; ++t) {
      const v8h* p0 = (const v8h*)(&lds_h[wave][col][t * 32 + kq * 8]);
      const v8h* p1 = (const v8h*)(&lds_h[wave][col][t * 32 + 16 + kq * 8]);
      v8h lo = *p0, hi = *p1;
      #pragma unroll
      for (int j = 0; j < 8; ++j) { ha[t][j] = lo[j]; ha[t][8 + j] = hi[j]; }
    }

    // ---- layer 2: out = h @ W2 + b2, 2 WMMAs -----------------------------
    v8f o = {};
    o = wmma_f16(ha[0], bw2[0], o);
    o = wmma_f16(ha[1], bw2[1], o);

    #pragma unroll
    for (int rr = 0; rr < 8; ++rr) {
      const int erow = base + kq * 8 + rr;
      if (erow < E)
        out[(size_t)erow * EDGE_OUT + col] = o[rr] + b2v;
    }
  }
}

extern "C" void kernel_launch(void* const* d_in, const int* in_sizes, int n_in,
                              void* d_out, int out_size, void* d_ws, size_t ws_size,
                              hipStream_t stream) {
  const float* src  = (const float*)d_in[0];
  const float* dst  = (const float*)d_in[1];
  const float* ea   = (const float*)d_in[2];
  const float* u    = (const float*)d_in[3];
  const int*   bat  = (const int*)  d_in[4];
  const float* W1   = (const float*)d_in[5];
  const float* b1   = (const float*)d_in[6];
  const float* W2   = (const float*)d_in[7];
  const float* b2   = (const float*)d_in[8];
  float* out = (float*)d_out;

  const int E = in_sizes[0] / NODE_IN;
  const int n_tiles = (E + 15) / 16;
  int waves_needed = (n_tiles + TILES_PER_WAVE - 1) / TILES_PER_WAVE;
  int blocks = (waves_needed + WAVES_PER_BLOCK - 1) / WAVES_PER_BLOCK;
  if (blocks < 1) blocks = 1;

  edge_mlp_wmma<<<blocks, BLOCK_THREADS, 0, stream>>>(
      src, dst, ea, u, bat, W1, b1, W2, b2, out, E, n_tiles);
}
